// GWNet_51728586113698
// MI455X (gfx1250) — matrix operand order repaired and checked by reading, not compiled, not run on hardware
//
#include <hip/hip_runtime.h>
#include <hip/hip_bf16.h>
#include <stdint.h>

#define N_NODES 8192
#define NNZ     131072

typedef __attribute__((ext_vector_type(16))) __bf16 v16bf;
typedef __attribute__((ext_vector_type(8)))  float  v8f;

union FragU  { v16bf v; uint4 q[2]; };
union U16x8  { uint4 q; uint16_t s[8]; };

__device__ __forceinline__ uint16_t f2bf(float x) {
    uint32_t u = __float_as_uint(x);
    u += 0x7FFFu + ((u >> 16) & 1u);           // round-to-nearest-even
    return (uint16_t)(u >> 16);
}

__device__ __forceinline__ v8f vzero8() {
    v8f v;
#pragma unroll
    for (int i = 0; i < 8; ++i) v[i] = 0.0f;
    return v;
}

__device__ __forceinline__ v8f wmma_bf16(v16bf a, v16bf b, v8f c) {
    return __builtin_amdgcn_wmma_f32_16x16x32_bf16(
        /*neg_a=*/false, a, /*neg_b=*/false, b,
        /*c_mod=*/(short)0, c, /*reuse_a=*/false, /*reuse_b=*/false);
}

// A fragment (16x32 bf16, MxK): lane r=l&15 row, h=l>>4; K chunks {h*8..+7, 16+h*8..+7}
__device__ __forceinline__ v16bf load_frag_A(const uint16_t* s, int row0, int stride, int k0, int lane) {
    int r = lane & 15, h = lane >> 4;
    const uint16_t* p = s + (size_t)(row0 + r) * stride + k0 + h * 8;
    FragU u;
    u.q[0] = *(const uint4*)(p);
    u.q[1] = *(const uint4*)(p + 16);
    return u.v;
}

// B fragment (32x16 bf16, KxN), from column-as-LDS-row storage: lane c=l&15 col, h=l>>4;
// K chunks {h*8..+7, 16+h*8..+7}
__device__ __forceinline__ v16bf load_frag_B(const uint16_t* s, int col0, int stride, int k0, int lane) {
    int c = lane & 15, h = lane >> 4;
    const uint16_t* p = s + (size_t)(col0 + c) * stride + k0 + h * 8;
    FragU u;
    u.q[0] = *(const uint4*)(p);
    u.q[1] = *(const uint4*)(p + 16);
    return u.v;
}

// ---------------- transpose X (512 x 8192) -> Xt (8192 x 512) ----------------
__global__ void transpose_x_kernel(const float* __restrict__ X, float* __restrict__ Xt) {
    int gid = blockIdx.x * 256 + threadIdx.x;       // 8192*512 threads
    int f = gid & 511, i = gid >> 9;
    Xt[(size_t)i * 512 + f] = X[(size_t)f * N_NODES + i];
}

// ---------------- colsum[j] = sum_i exp(relu(Z Z^T)[i,j]) via WMMA ----------------
__global__ __launch_bounds__(256) void zz_colsum_kernel(const float* __restrict__ Z,
                                                        float* __restrict__ colsum) {
    __shared__ uint16_t Zi[128 * 72];
    __shared__ uint16_t Zj[128 * 72];
    __shared__ float colpart[128];
    const int tid = threadIdx.x, lane = tid & 31, wave = tid >> 5;
    const int wm = wave & 3, wn = wave >> 2, hl = lane >> 4, nl = lane & 15;
    const int ib = blockIdx.x * 128, jb = blockIdx.y * 128;

    for (int c = tid; c < 2048; c += 256) {
        int row = c >> 4, q = c & 15;
        float4 vi = *(const float4*)(Z + (size_t)(ib + row) * 64 + q * 4);
        float4 vj = *(const float4*)(Z + (size_t)(jb + row) * 64 + q * 4);
        ushort4 si; si.x = f2bf(vi.x); si.y = f2bf(vi.y); si.z = f2bf(vi.z); si.w = f2bf(vi.w);
        ushort4 sj; sj.x = f2bf(vj.x); sj.y = f2bf(vj.y); sj.z = f2bf(vj.z); sj.w = f2bf(vj.w);
        *(ushort4*)(Zi + (size_t)row * 72 + q * 4) = si;
        *(ushort4*)(Zj + (size_t)row * 72 + q * 4) = sj;
    }
    if (tid < 128) colpart[tid] = 0.0f;
    __syncthreads();

    v8f ez[2][4];
#pragma unroll
    for (int sm = 0; sm < 2; ++sm)
#pragma unroll
        for (int sn = 0; sn < 4; ++sn) ez[sm][sn] = vzero8();

#pragma unroll
    for (int k0 = 0; k0 < 64; k0 += 32) {
        v16bf a0 = load_frag_A(Zi, wm * 32 + 0, 72, k0, lane);
        v16bf a1 = load_frag_A(Zi, wm * 32 + 16, 72, k0, lane);
#pragma unroll
        for (int sn = 0; sn < 4; ++sn) {
            v16bf b = load_frag_B(Zj, wn * 64 + sn * 16, 72, k0, lane);
            ez[0][sn] = wmma_bf16(a0, b, ez[0][sn]);
            ez[1][sn] = wmma_bf16(a1, b, ez[1][sn]);
        }
    }

    float psum[4] = {0.f, 0.f, 0.f, 0.f};
#pragma unroll
    for (int sm = 0; sm < 2; ++sm)
#pragma unroll
        for (int sn = 0; sn < 4; ++sn)
#pragma unroll
            for (int r = 0; r < 8; ++r)
                psum[sn] += __expf(fmaxf(ez[sm][sn][r], 0.0f));
#pragma unroll
    for (int sn = 0; sn < 4; ++sn)
        atomicAdd(&colpart[wn * 64 + sn * 16 + nl], psum[sn]);
    __syncthreads();
    if (tid < 128) atomicAdd(&colsum[jb + tid], colpart[tid]);
}

// ---------------- Xs = bf16(Xt / colsum[row]) ----------------
__global__ void scale_bf16_kernel(const float* __restrict__ Xt, const float* __restrict__ colsum,
                                  uint16_t* __restrict__ Xs, int F) {
    int gid = blockIdx.x * 256 + threadIdx.x;       // n*F threads
    int j = gid / F;
    Xs[gid] = f2bf(Xt[gid] / colsum[j]);
}

// ---------------- edge-list SpMM with global f32 atomics ----------------
__global__ void spmm_kernel(const int* __restrict__ idx, const float* __restrict__ vals,
                            const float* __restrict__ Xin, float* __restrict__ Yout,
                            int F, float scale) {
    int gid = blockIdx.x * 256 + threadIdx.x;       // NNZ * (F/4) threads
    int nch = F >> 2;
    int e = gid / nch, c = gid % nch;
    int rdst = idx[e];
    int rsrc = idx[NNZ + e];
    float v = vals[e] * scale;
    float4 x = *(const float4*)(Xin + (size_t)rsrc * F + c * 4);
    float* y = Yout + (size_t)rdst * F + c * 4;
    atomicAdd(y + 0, v * x.x);
    atomicAdd(y + 1, v * x.y);
    atomicAdd(y + 2, v * x.z);
    atomicAdd(y + 3, v * x.w);
}

// ---- fused: Tacc[i,f] = Tacc + X1a + X1b - Xt + sum_j exp(relu(ZZ^T))[i,j] * Xs[j,f] ----
__global__ __launch_bounds__(256) void fused_attn_kernel(const float* __restrict__ Z,
                                                         const uint16_t* __restrict__ Xs,
                                                         const float* __restrict__ X1a,
                                                         const float* __restrict__ X1b,
                                                         const float* __restrict__ Xt,
                                                         float* __restrict__ Tacc, int F) {
    __shared__ uint16_t Zi[128 * 72];
    __shared__ uint16_t Zj[128 * 72];
    __shared__ uint16_t Et[128 * 136];
    __shared__ uint16_t Xsb[128 * 40];
    const int tid = threadIdx.x, lane = tid & 31, wave = tid >> 5;
    const int wm = wave & 3, wn = wave >> 2, hl = lane >> 4, nl = lane & 15;
    const int ib = blockIdx.x * 128, fb = blockIdx.y * 128;

    for (int c = tid; c < 2048; c += 256) {                 // Zi: rows ib..ib+127, once
        int row = c >> 4, q = c & 15;
        float4 v = *(const float4*)(Z + (size_t)(ib + row) * 64 + q * 4);
        ushort4 s; s.x = f2bf(v.x); s.y = f2bf(v.y); s.z = f2bf(v.z); s.w = f2bf(v.w);
        *(ushort4*)(Zi + (size_t)row * 72 + q * 4) = s;
    }

    v8f acc[2][4];
#pragma unroll
    for (int sm = 0; sm < 2; ++sm)
#pragma unroll
        for (int sn = 0; sn < 4; ++sn) acc[sm][sn] = vzero8();

    for (int jb = 0; jb < N_NODES; jb += 128) {
        __syncthreads();
        for (int c = tid; c < 2048; c += 256) {             // Zj: rows jb..jb+127
            int row = c >> 4, q = c & 15;
            float4 v = *(const float4*)(Z + (size_t)(jb + row) * 64 + q * 4);
            ushort4 s; s.x = f2bf(v.x); s.y = f2bf(v.y); s.z = f2bf(v.z); s.w = f2bf(v.w);
            *(ushort4*)(Zj + (size_t)row * 72 + q * 4) = s;
        }
        __syncthreads();

        v8f ez[2][4];
#pragma unroll
        for (int sm = 0; sm < 2; ++sm)
#pragma unroll
            for (int sn = 0; sn < 4; ++sn) ez[sm][sn] = vzero8();
#pragma unroll
        for (int k0 = 0; k0 < 64; k0 += 32) {
            v16bf a0 = load_frag_A(Zi, wm * 32 + 0, 72, k0, lane);
            v16bf a1 = load_frag_A(Zi, wm * 32 + 16, 72, k0, lane);
#pragma unroll
            for (int sn = 0; sn < 4; ++sn) {
                v16bf b = load_frag_B(Zj, wn * 64 + sn * 16, 72, k0, lane);
                ez[0][sn] = wmma_bf16(a0, b, ez[0][sn]);
                ez[1][sn] = wmma_bf16(a1, b, ez[1][sn]);
            }
        }
        // E tile -> LDS (bf16), rows = local i, cols = local j
#pragma unroll
        for (int sm = 0; sm < 2; ++sm)
#pragma unroll
            for (int sn = 0; sn < 4; ++sn) {
                int il = wm * 32 + sm * 16 + 8 * hl;
                int jl = wn * 64 + sn * 16 + nl;
#pragma unroll
                for (int r = 0; r < 8; ++r)
                    Et[(size_t)(il + r) * 136 + jl] = f2bf(__expf(fmaxf(ez[sm][sn][r], 0.0f)));
            }
        __syncthreads();

#pragma unroll 1
        for (int kc = 0; kc < 4; ++kc) {
            for (int c = tid; c < 512; c += 256) {          // Xs chunk, transposed into Xsb
                int kk = c >> 4, c8 = c & 15;
                const uint16_t* src = Xs + (size_t)(jb + kc * 32 + kk) * F + fb + c8 * 8;
                U16x8 u; u.q = *(const uint4*)src;
#pragma unroll
                for (int t = 0; t < 8; ++t)
                    Xsb[(size_t)(c8 * 8 + t) * 40 + kk] = u.s[t];
            }
            __syncthreads();
            v16bf ea0 = load_frag_A(Et, wm * 32 + 0, 136, kc * 32, lane);
            v16bf ea1 = load_frag_A(Et, wm * 32 + 16, 136, kc * 32, lane);
#pragma unroll
            for (int sn = 0; sn < 4; ++sn) {
                v16bf xb = load_frag_B(Xsb, wn * 64 + sn * 16, 40, 0, lane);
                acc[0][sn] = wmma_bf16(ea0, xb, acc[0][sn]);
                acc[1][sn] = wmma_bf16(ea1, xb, acc[1][sn]);
            }
            __syncthreads();
        }
    }

    // epilogue: combine Chebyshev/spmm terms
#pragma unroll
    for (int sm = 0; sm < 2; ++sm)
#pragma unroll
        for (int sn = 0; sn < 4; ++sn)
#pragma unroll
            for (int r = 0; r < 8; ++r) {
                int gi = ib + wm * 32 + sm * 16 + 8 * hl + r;
                int gf = fb + wn * 64 + sn * 16 + nl;
                size_t off = (size_t)gi * F + gf;
                Tacc[off] = Tacc[off] + acc[sm][sn][r] + X1a[off] + X1b[off] - Xt[off];
            }
}

// ---- layer-1 head: Xt2[i, b*128+o] = relu( sum_k total[i, b*64+k] * W1[k,o] ) ----
__global__ void head1_kernel(const float* __restrict__ total, const float* __restrict__ W,
                             float* __restrict__ Xt2) {
    int gid = blockIdx.x * 256 + threadIdx.x;       // 8192*8*128 threads
    int o = gid & 127, b = (gid >> 7) & 7, i = gid >> 10;
    const float* t = total + (size_t)i * 512 + b * 64;
    float s = 0.0f;
#pragma unroll
    for (int k = 0; k < 64; ++k) s += t[k] * W[k * 128 + o];
    Xt2[(size_t)i * 1024 + b * 128 + o] = fmaxf(s, 0.0f);
}

// ---- layer-2 head + node mean: out[b,o] = mean_i relu( sum_k total[i,b*128+k] * W2[k,o] ) ----
__global__ void head2_kernel(const float* __restrict__ total, const float* __restrict__ W,
                             float* __restrict__ out) {
    __shared__ float red[256];
    int b = blockIdx.x >> 7, o = blockIdx.x & 127, tid = threadIdx.x;
    float p = 0.0f;
    for (int i = tid; i < N_NODES; i += 256) {
        const float* t = total + (size_t)i * 1024 + b * 128;
        float s = 0.0f;
#pragma unroll
        for (int k = 0; k < 128; ++k) s += t[k] * W[k * 128 + o];
        p += fmaxf(s, 0.0f);
    }
    red[tid] = p;
    __syncthreads();
    for (int st = 128; st > 0; st >>= 1) {
        if (tid < st) red[tid] += red[tid + st];
        __syncthreads();
    }
    if (tid == 0) out[blockIdx.x] = red[0] * (1.0f / (float)N_NODES);
}

extern "C" void kernel_launch(void* const* d_in, const int* in_sizes, int n_in,
                              void* d_out, int out_size, void* d_ws, size_t ws_size,
                              hipStream_t stream) {
    (void)in_sizes; (void)n_in; (void)out_size; (void)ws_size;
    const int*   A1idx = (const int*)d_in[0];
    const float* A1val = (const float*)d_in[1];
    const int*   A2idx = (const int*)d_in[2];
    const float* A2val = (const float*)d_in[3];
    const float* X     = (const float*)d_in[4];
    const float* Z     = (const float*)d_in[5];
    const float* W1    = (const float*)d_in[6];
    const float* W2    = (const float*)d_in[7];
    float* out = (float*)d_out;

    // workspace layout
    char* ws = (char*)d_ws;
    size_t off = 0;
    float*    colsum = (float*)(ws + off);    off += (size_t)N_NODES * 4;            // 32 KB
    float*    Xt     = (float*)(ws + off);    off += (size_t)N_NODES * 1024 * 4;     // 32 MB
    uint16_t* Xs     = (uint16_t*)(ws + off); off += (size_t)N_NODES * 1024 * 2;     // 16 MB
    float*    X1a    = (float*)(ws + off);    off += (size_t)N_NODES * 1024 * 4;     // 32 MB
    float*    X1b    = (float*)(ws + off);    off += (size_t)N_NODES * 1024 * 4;     // 32 MB
    float*    Tacc   = (float*)(ws + off);    off += (size_t)N_NODES * 1024 * 4;     // 32 MB

    // once: transpose input, softmax column denominators
    hipMemsetAsync(colsum, 0, (size_t)N_NODES * 4, stream);
    transpose_x_kernel<<<(N_NODES * 512) / 256, 256, 0, stream>>>(X, Xt);
    zz_colsum_kernel<<<dim3(N_NODES / 128, N_NODES / 128), 256, 0, stream>>>(Z, colsum);

    for (int layer = 0; layer < 2; ++layer) {
        const int F = layer ? 1024 : 512;
        const size_t bytes = (size_t)N_NODES * F * 4;
        hipMemsetAsync(X1a, 0, bytes, stream);
        hipMemsetAsync(X1b, 0, bytes, stream);
        hipMemsetAsync(Tacc, 0, bytes, stream);

        const int spmm_blocks = NNZ * (F / 4) / 256;
        spmm_kernel<<<spmm_blocks, 256, 0, stream>>>(A1idx, A1val, Xt, X1a, F, 1.0f);
        spmm_kernel<<<spmm_blocks, 256, 0, stream>>>(A2idx, A2val, Xt, X1b, F, 1.0f);
        spmm_kernel<<<spmm_blocks, 256, 0, stream>>>(A1idx, A1val, X1a, Tacc, F, 2.0f);
        spmm_kernel<<<spmm_blocks, 256, 0, stream>>>(A2idx, A2val, X1b, Tacc, F, 2.0f);

        scale_bf16_kernel<<<(N_NODES * F) / 256, 256, 0, stream>>>(Xt, colsum, Xs, F);

        fused_attn_kernel<<<dim3(N_NODES / 128, F / 128), 256, 0, stream>>>(
            Z, Xs, X1a, X1b, Xt, Tacc, F);

        if (layer == 0)
            head1_kernel<<<(N_NODES * 8 * 128) / 256, 256, 0, stream>>>(Tacc, W1, Xt);
        else
            head2_kernel<<<8 * 128, 256, 0, stream>>>(Tacc, W2, out);
    }
}
